// basicGNN_21105469293031
// MI455X (gfx1250) — matrix operand order repaired and checked by reading, not compiled, not run on hardware
//
#include <hip/hip_runtime.h>
#include <hip/hip_bf16.h>
#include <stdint.h>

#define DIM   512
#define BLK_M 128
#define BLK_N 64
#define BLK_K 64
#define LDA_P 72   // 64 bf16 row + 4-dword TDM pad -> 144B pitch (16B aligned)
#define LDB_P 72
#define SMA   (BLK_M * LDA_P)          // A tile elems per buffer
#define SMB   (BLK_N * LDB_P)          // B tile elems per buffer
#define SMBUF (SMA + SMB)              // elems per double-buffer stage

typedef __attribute__((ext_vector_type(16))) __bf16   v16bf;
typedef __attribute__((ext_vector_type(8)))  __bf16   v8bf;
typedef __attribute__((ext_vector_type(8)))  float    v8f;
typedef __attribute__((ext_vector_type(4)))  uint32_t u32x4;
typedef __attribute__((ext_vector_type(8)))  uint32_t u32x8;

__device__ __forceinline__ __bf16 to_bf16(float f) {
    uint32_t u = __builtin_bit_cast(uint32_t, f);
    uint32_t r = u + 0x7FFFu + ((u >> 16) & 1u);     // RNE
    uint16_t h = (uint16_t)(r >> 16);
    return __builtin_bit_cast(__bf16, h);
}

// ---------------------------------------------------------------------------
// TDM: DMA a 2-D bf16 tile (tile_x x tile_y elems) from global into LDS.
// Descriptor per CDNA5 ISA ch.8; dims are relative to the tile start so that
// rows beyond tensor_y zero-fill. Row padding: every 2^(pi+1) dwords stored,
// insert (pa+1) dwords -> LDS pitch = tile_x + pad elems.
// ---------------------------------------------------------------------------
__device__ __forceinline__ void tdm_load_2d_bf16(
        uint32_t lds_byte_off, const __bf16* gptr,
        uint32_t tile_x, uint32_t tile_y,
        uint32_t tensor_x, uint32_t tensor_y,
        uint32_t stride_x_elems,
        uint32_t pad_interval_code, uint32_t pad_amount_code)
{
    uint64_t ga = (uint64_t)(uintptr_t)gptr;
    u32x4 g0;
    g0[0] = 0x1u;                                        // count=1, user D#
    g0[1] = lds_byte_off;                                // lds_addr
    g0[2] = (uint32_t)ga;                                // global_addr[31:0]
    g0[3] = (uint32_t)((ga >> 32) & 0x1FFFFFFu)          // global_addr[56:32]
          | (2u << 30);                                  // type=2 ("image")
    u32x8 g1;
    g1[0] = (1u << 16)                                   // data_size=1 -> 2B
          | (1u << 20)                                   // pad_enable
          | (pad_interval_code << 22)
          | (pad_amount_code << 25);
    g1[1] = (tensor_x & 0xFFFFu) << 16;                  // tensor_dim0[15:0]
    g1[2] = (tensor_x >> 16) | ((tensor_y & 0xFFFFu) << 16); // dim0 hi | dim1 lo
    g1[3] = (tensor_y >> 16) | (tile_x << 16);           // dim1 hi | tile_dim0
    g1[4] = tile_y;                                      // tile_dim1 ; tile_dim2=0
    g1[5] = stride_x_elems;                              // tensor_dim0_stride
    g1[6] = 0;
    g1[7] = 0;
    asm volatile("tensor_load_to_lds %0, %1" :: "s"(g0), "s"(g1) : "memory");
}

// ---------------------------------------------------------------------------
// C[N,512] = A_bf16[N,512] @ W  with W given pre-transposed bf16 WT[n][k].
// 256 thr = 8 waves; block tile 128x64; K-step 64; TDM double-buffered LDS.
// ---------------------------------------------------------------------------
__global__ __launch_bounds__(256)
void gemm_tdm_wmma(const __bf16* __restrict__ A, const __bf16* __restrict__ WT,
                   float* __restrict__ C, int n_rows) {
    __shared__ __attribute__((aligned(128))) __bf16 smem[2 * SMBUF];

    const int tid  = threadIdx.x;
    const int lane = tid & 31;
    const int wave = tid >> 5;                 // 0..7
    const int r0   = blockIdx.x * BLK_M;
    const int c0   = blockIdx.y * BLK_N;
    const int hlf  = lane >> 4;
    const int m    = lane & 15;

    const uint32_t lds_base = (uint32_t)(uintptr_t)&smem[0];
    const uint32_t remY_A   = (uint32_t)(n_rows - r0);   // rows left from tile start

    v8f acc[4] = {};

    // ---- prologue: stage tile kb=0 into buffer 0 ----
    if (wave == 0) {
        tdm_load_2d_bf16(lds_base,              A  + (size_t)r0 * DIM,
                         BLK_K, BLK_M, DIM, remY_A, DIM, 4, 3);
        tdm_load_2d_bf16(lds_base + SMA * 2,    WT + (size_t)c0 * DIM,
                         BLK_K, BLK_N, DIM, (uint32_t)(DIM - c0), DIM, 4, 3);
    }
    __builtin_amdgcn_s_wait_tensorcnt(0);
    __syncthreads();

    int cur = 0;
    for (int kb = 0; kb < DIM; kb += BLK_K) {
        // ---- issue DMA for next K tile into the other buffer ----
        if (kb + BLK_K < DIM && wave == 0) {
            uint32_t nb = (uint32_t)(cur ^ 1) * (SMBUF * 2);
            tdm_load_2d_bf16(lds_base + nb,           A  + (size_t)r0 * DIM + kb + BLK_K,
                             BLK_K, BLK_M, (uint32_t)(DIM - kb - BLK_K), remY_A, DIM, 4, 3);
            tdm_load_2d_bf16(lds_base + nb + SMA * 2, WT + (size_t)c0 * DIM + kb + BLK_K,
                             BLK_K, BLK_N, (uint32_t)(DIM - kb - BLK_K),
                             (uint32_t)(DIM - c0), DIM, 4, 3);
        }

        const __bf16* lA = smem + cur * SMBUF;
        const __bf16* lB = lA + SMA;

        #pragma unroll
        for (int s = 0; s < 2; ++s) {          // two K=32 sub-steps
            // A frag (ISA 16-bit A layout): lane<16 K=0..7,16..23 ; >=16 +8
            const __bf16* pa = &lA[(wave * 16 + m) * LDA_P + s * 32 + hlf * 8];
            v8bf a_lo = *(const v8bf*)pa;
            v8bf a_hi = *(const v8bf*)(pa + 16);
            v16bf afrag = __builtin_shufflevector(a_lo, a_hi,
                            0,1,2,3,4,5,6,7,8,9,10,11,12,13,14,15);
            #pragma unroll
            for (int t = 0; t < 4; ++t) {
                // B frag: lane n=m ; lane<16 K=0..7,16..23 ; >=16 +8
                const __bf16* pb = &lB[(t * 16 + m) * LDB_P + s * 32 + hlf * 8];
                v8bf b_lo = *(const v8bf*)pb;
                v8bf b_hi = *(const v8bf*)(pb + 16);
                v16bf bfrag = __builtin_shufflevector(b_lo, b_hi,
                                0,1,2,3,4,5,6,7,8,9,10,11,12,13,14,15);
                acc[t] = __builtin_amdgcn_wmma_f32_16x16x32_bf16(
                             false, afrag, false, bfrag,
                             (short)0, acc[t], false, false);
            }
        }

        __builtin_amdgcn_s_wait_tensorcnt(0);  // next-tile DMA landed
        __syncthreads();                       // + everyone done reading cur
        cur ^= 1;
    }

    // ---- epilogue: C/D layout: lane n = lane&15, rows e + 8*(lane>>4) ----
    #pragma unroll
    for (int t = 0; t < 4; ++t) {
        int col = c0 + t * 16 + m;
        #pragma unroll
        for (int e = 0; e < 8; ++e) {
            int row = r0 + wave * 16 + hlf * 8 + e;
            if (row < n_rows) C[(size_t)row * DIM + col] = acc[t][e];
        }
    }
}

// ---------------------------------------------------------------------------
// conversion pre-passes
// ---------------------------------------------------------------------------
// W[k][n] f32 -> WT[n][k] bf16 (512x512), LDS-tiled transpose
__global__ __launch_bounds__(256)
void convert_w_t(const float* __restrict__ W, __bf16* __restrict__ WT) {
    __shared__ float t[32][33];
    int bx = blockIdx.x * 32, by = blockIdx.y * 32;   // bx: n, by: k
    int tx = threadIdx.x & 31, ty = threadIdx.x >> 5; // ty 0..7
    #pragma unroll
    for (int i = 0; i < 32; i += 8)
        t[ty + i][tx] = W[(size_t)(by + ty + i) * DIM + bx + tx];
    __syncthreads();
    #pragma unroll
    for (int i = 0; i < 32; i += 8)
        WT[(size_t)(bx + ty + i) * DIM + by + tx] = to_bf16(t[tx][ty + i]);
}

// f32 -> bf16 (8 elems/thread), optional fused ReLU
__global__ __launch_bounds__(256)
void convert_bf16(const float* __restrict__ src, __bf16* __restrict__ dst,
                  int n, int relu) {
    int i = (blockIdx.x * blockDim.x + threadIdx.x) << 3;
    if (i >= n) return;
    float4 a = *(const float4*)&src[i];
    float4 b = *(const float4*)&src[i + 4];
    if (relu) {
        a.x = fmaxf(a.x, 0.f); a.y = fmaxf(a.y, 0.f);
        a.z = fmaxf(a.z, 0.f); a.w = fmaxf(a.w, 0.f);
        b.x = fmaxf(b.x, 0.f); b.y = fmaxf(b.y, 0.f);
        b.z = fmaxf(b.z, 0.f); b.w = fmaxf(b.w, 0.f);
    }
    v8bf o;
    o[0] = to_bf16(a.x); o[1] = to_bf16(a.y); o[2] = to_bf16(a.z); o[3] = to_bf16(a.w);
    o[4] = to_bf16(b.x); o[5] = to_bf16(b.y); o[6] = to_bf16(b.z); o[7] = to_bf16(b.w);
    *(v8bf*)&dst[i] = o;
}

// ---------------------------------------------------------------------------
// degree / rsqrt-normalization
// ---------------------------------------------------------------------------
__global__ void deg_init(float* deg, int n) {
    int i = blockIdx.x * blockDim.x + threadIdx.x;
    if (i < n) deg[i] = 1.0f;                    // self-loop
}
__global__ void deg_scatter(const int* __restrict__ dst, float* deg, int E) {
    int e = blockIdx.x * blockDim.x + threadIdx.x;
    if (e < E) atomicAdd(&deg[dst[e]], 1.0f);
}
__global__ void deg_finalize(float* deg, int n) {
    int i = blockIdx.x * blockDim.x + threadIdx.x;
    if (i < n) deg[i] = rsqrtf(deg[i]);          // deg >= 1 always
}

// out[i,:] = h[i,:] * dinv[i]^2 + bias[:]       (self-loop term + bias)
__global__ __launch_bounds__(256)
void agg_init(const float* __restrict__ h, const float* __restrict__ dinv,
              const float* __restrict__ bias, float* __restrict__ out, int n_rows) {
    int idx = blockIdx.x * blockDim.x + threadIdx.x;
    int e4 = idx << 2;
    if (e4 >= n_rows * DIM) return;
    int row = e4 >> 9;
    int col = e4 & (DIM - 1);
    float di = dinv[row];
    float s = di * di;
    float4 v = *(const float4*)&h[e4];
    float4 b = *(const float4*)&bias[col];
    float4 o;
    o.x = v.x * s + b.x; o.y = v.y * s + b.y;
    o.z = v.z * s + b.z; o.w = v.w * s + b.w;
    *(float4*)&out[e4] = o;
}

// one wave per edge: out[d,:] += h[s,:] * dinv[s]*dinv[d]  (atomics hit L2)
__global__ __launch_bounds__(256)
void agg_edges(const float* __restrict__ h, const int* __restrict__ src,
               const int* __restrict__ dst, const float* __restrict__ dinv,
               float* __restrict__ out, int E) {
    int wid  = blockIdx.x * 8 + (threadIdx.x >> 5);
    int lane = threadIdx.x & 31;
    if (wid >= E) return;
    int s = src[wid], d = dst[wid];
    float norm = dinv[s] * dinv[d];
    const float* hs = &h[(size_t)s * DIM];
    float* od = &out[(size_t)d * DIM];
    #pragma unroll
    for (int j = 0; j < 16; ++j) {
        int c = lane + j * 32;                   // coalesced within wave
        atomicAdd(&od[c], hs[c] * norm);
    }
}

// column mean: u[c] = sum_r h[r,c] / n_rows ; 8 blocks x 64 cols
__global__ __launch_bounds__(256)
void col_mean(const float* __restrict__ h, float* __restrict__ u, int n_rows) {
    __shared__ float red[4][64];
    int t  = threadIdx.x;
    int c  = (blockIdx.x << 6) + (t & 63);
    int rg = t >> 6;
    float s = 0.f;
    for (int r = rg; r < n_rows; r += 4) s += h[(size_t)r * DIM + c];
    red[rg][t & 63] = s;
    __syncthreads();
    if (t < 64) {
        float tot = red[0][t] + red[1][t] + red[2][t] + red[3][t];
        u[(blockIdx.x << 6) + t] = tot / (float)n_rows;
    }
}

// head: z=[u1,u2] (1024) -> relu(z@Wf1+bf1) (512) -> sigmoid(.@Wf2+bf2)
__global__ __launch_bounds__(512)
void head_mlp(const float* __restrict__ u, const float* __restrict__ Wf1,
              const float* __restrict__ bf1, const float* __restrict__ Wf2,
              const float* __restrict__ bf2, float* __restrict__ out) {
    __shared__ float red[512];
    int j = threadIdx.x;
    float s = bf1[j];
    for (int i = 0; i < 2 * DIM; ++i) s += u[i] * Wf1[(size_t)i * DIM + j];
    float hj = fmaxf(s, 0.f);
    red[j] = hj * Wf2[j];
    __syncthreads();
    for (int off = 256; off > 0; off >>= 1) {
        if (j < off) red[j] += red[j + off];
        __syncthreads();
    }
    if (j == 0) {
        float z = red[0] + bf2[0];
        out[0] = 1.0f / (1.0f + expf(-z));
    }
}

// ---------------------------------------------------------------------------
extern "C" void kernel_launch(void* const* d_in, const int* in_sizes, int n_in,
                              void* d_out, int out_size, void* d_ws, size_t ws_size,
                              hipStream_t stream) {
    const int N = in_sizes[0] / DIM;             // 10000
    const int E = in_sizes[2] / 2;               // 160000

    const float* x[2]     = { (const float*)d_in[0], (const float*)d_in[1] };
    const int*   edges[2] = { (const int*)d_in[2],   (const int*)d_in[3]   };
    const float* Wl[3]    = { (const float*)d_in[4], (const float*)d_in[5], (const float*)d_in[6] };
    const float* bl[3]    = { (const float*)d_in[7], (const float*)d_in[8], (const float*)d_in[9] };
    const float* Wf1 = (const float*)d_in[10];
    const float* bf1 = (const float*)d_in[11];
    const float* Wf2 = (const float*)d_in[12];
    const float* bf2 = (const float*)d_in[13];

    float*  bufA = (float*)d_ws;                 // [N*DIM] gemm output
    float*  bufB = bufA + (size_t)N * DIM;       // [N*DIM] aggregated output
    float*  dinv = bufB + (size_t)N * DIM;       // [N]
    float*  u    = dinv + N;                     // [2*DIM] (u1 | u2)
    __bf16* Abf  = (__bf16*)(u + 2 * DIM);       // [N*DIM]  bf16 activations
    __bf16* WT   = Abf + (size_t)N * DIM;        // [3*DIM*DIM] bf16 W^T

    const dim3 gemm_grid((N + BLK_M - 1) / BLK_M, DIM / BLK_N);
    const int  n_el   = N * DIM;
    const int  cvt_bl = (n_el / 8 + 255) / 256;

    // one-time: W -> bf16 transposed
    for (int l = 0; l < 3; ++l)
        convert_w_t<<<dim3(16, 16), 256, 0, stream>>>(Wl[l], WT + (size_t)l * DIM * DIM);

    for (int g = 0; g < 2; ++g) {
        deg_init    <<<(N + 255) / 256, 256, 0, stream>>>(dinv, N);
        deg_scatter <<<(E + 255) / 256, 256, 0, stream>>>(edges[g] + E, dinv, E);
        deg_finalize<<<(N + 255) / 256, 256, 0, stream>>>(dinv, N);

        const float* in = x[g];
        for (int l = 0; l < 3; ++l) {
            convert_bf16<<<cvt_bl, 256, 0, stream>>>(in, Abf, n_el, l > 0 ? 1 : 0);
            gemm_tdm_wmma<<<gemm_grid, 256, 0, stream>>>(Abf, WT + (size_t)l * DIM * DIM, bufA, N);
            agg_init <<<(n_el / 4 + 255) / 256, 256, 0, stream>>>(bufA, dinv, bl[l], bufB, N);
            agg_edges<<<(E + 7) / 8, 256, 0, stream>>>(bufA, edges[g], edges[g] + E, dinv, bufB, E);
            in = bufB;                            // ReLU folded into next convert
        }
        col_mean<<<DIM / 64, 256, 0, stream>>>(bufB, u + (size_t)g * DIM, N);
    }

    head_mlp<<<1, 512, 0, stream>>>(u, Wf1, bf1, Wf2, bf2, (float*)d_out);
}
